// SelfAttention_137438954165
// MI455X (gfx1250) — matrix operand order repaired and checked by reading, not compiled, not run on hardware
//
#include <hip/hip_runtime.h>

// ---------------------------------------------------------------------------
// Causal self-attention forward for MI455X (gfx1250, wave32, WMMA, TDM).
// [0a] x -> bf16, [0b] w^T -> bf16 (one-time prep: hoists all convert/transpose
//      out of the GEMM hot loops)
// [1]  QKV GEMM (bf16 WMMA, f32 acc), TDM double-buffered LDS staging
// [2]  transposed flash attention (S^T = K Q^T, O^T = V^T P^T), TDM staging
// [3]  output GEMM + bias -> fp32, TDM staging
// ---------------------------------------------------------------------------

#define D_MODEL  1024
#define N_HEAD   16
#define HEAD_DIM 64
#define T_SEQ    2048
#define BATCH    2
#define M_TOK    (BATCH * T_SEQ)   // 4096

typedef __attribute__((ext_vector_type(16))) __bf16 bf16x16;
typedef __attribute__((ext_vector_type(8)))  __bf16 bf16x8;
typedef __attribute__((ext_vector_type(4)))  __bf16 bf16x4;
typedef __attribute__((ext_vector_type(8)))  float  f32x8;

typedef unsigned int       u32;
typedef unsigned long long u64;
typedef __attribute__((ext_vector_type(4))) u32 u32x4;
typedef __attribute__((ext_vector_type(8))) int i32x8;
typedef __attribute__((ext_vector_type(4))) int i32x4;

union AFragU { bf16x16 v; bf16x8 h[2]; };

__device__ __forceinline__ f32x8 zero8() {
  f32x8 z;
#pragma unroll
  for (int i = 0; i < 8; ++i) z[i] = 0.0f;
  return z;
}

__device__ __forceinline__ f32x8 wmma_bf16(bf16x16 a, bf16x16 b, f32x8 c) {
  return __builtin_amdgcn_wmma_f32_16x16x32_bf16(false, a, false, b, (short)0, c,
                                                 false, false);
}

// ---------------------------------------------------------------------------
// Tensor Data Mover staging (guarded; falls back to synchronous copies).
// ---------------------------------------------------------------------------
#if __has_builtin(__builtin_amdgcn_tensor_load_to_lds) && \
    __has_builtin(__builtin_amdgcn_s_wait_tensorcnt)
#define HAVE_TDM 1
#else
#define HAVE_TDM 0
#endif

// Low 32 bits of a generic shared-aperture pointer == LDS byte address.
__device__ __forceinline__ u32 lds_lo32(const void* p) {
  return (u32)(uintptr_t)p;
}

#if HAVE_TDM
// 2D tile load, bf16 (data_size=2B), LDS row padding via pad_interval/amount.
//   pint: pad after every 2^(pint+1) DWORDs stored; pamt: pad of (pamt+1) DWORDs.
// D# packing per CDNA5 ISA 8.3 (group0) / 8.4 (group1); groups 2,3 unused (2D).
__device__ __forceinline__ void tdm_load_2d(u32 lds_addr, const void* gtile,
                                            u32 tile_d0, u32 tile_d1,
                                            u32 stride0, u32 tdim0, u32 tdim1,
                                            u32 pint, u32 pamt)
{
  const u64 ga = (u64)(uintptr_t)gtile;
  u32x4 g0;
  g0[0] = 1u;                               // count=1 (valid user descriptor)
  g0[1] = lds_addr;                         // lds_addr[31:0] (bytes)
  g0[2] = (u32)ga;                          // global_addr[31:0] (tile start)
  g0[3] = (u32)(ga >> 32) | (2u << 30);     // global_addr[56:32] | type=2
  i32x8 g1;
  g1[0] = (int)((1u << 16) |                // data_size = 1 (2 bytes)
                (1u << 20) |                // pad_enable
                (pint << 22) | (pamt << 25));
  g1[1] = (int)((tdim0 & 0xFFFFu) << 16);                       // tensor_dim0 lo
  g1[2] = (int)((tdim0 >> 16) | ((tdim1 & 0xFFFFu) << 16));     // dim0 hi|dim1 lo
  g1[3] = (int)((tdim1 >> 16) | (tile_d0 << 16));               // dim1 hi|tile0
  g1[4] = (int)tile_d1;                                         // tile1 | tile2=0
  g1[5] = (int)stride0;                                         // dim0_stride lo
  g1[6] = 0;                                                    // stride hi / s1
  g1[7] = 0;
  const i32x4 z4 = {0, 0, 0, 0};
#if defined(__clang_major__) && (__clang_major__ >= 23)
  const i32x8 z8 = {0, 0, 0, 0, 0, 0, 0, 0};
  __builtin_amdgcn_tensor_load_to_lds(g0, g1, z4, z4, z8, 0);
#else
  __builtin_amdgcn_tensor_load_to_lds(g0, g1, z4, z4, 0);
#endif
}
#define TDM_WAIT(imm)                                           \
  do {                                                          \
    if ((threadIdx.x >> 5) == 0) __builtin_amdgcn_s_wait_tensorcnt(imm); \
  } while (0)
#else
#define TDM_WAIT(imm) ((void)0)
#endif

// ---------------------------------------------------------------------------
// Kernel 0a: x (fp32) -> xb (bf16), elementwise.
// ---------------------------------------------------------------------------
__global__ void __launch_bounds__(256)
cvt_x_kernel(const float* __restrict__ x, __bf16* __restrict__ xb)
{
  const size_t i = ((size_t)blockIdx.x * 256 + threadIdx.x) * 4;
  const float4 f = *(const float4*)(x + i);
  bf16x4 b4;
  b4[0] = (__bf16)f.x; b4[1] = (__bf16)f.y; b4[2] = (__bf16)f.z; b4[3] = (__bf16)f.w;
  *(bf16x4*)(xb + i) = b4;
}

// ---------------------------------------------------------------------------
// Kernel 0b: w [K][N] fp32 -> wt [N][K] bf16 (LDS-tiled 32x32 transpose).
// ---------------------------------------------------------------------------
__global__ void __launch_bounds__(256)
wtrans_kernel(const float* __restrict__ w, __bf16* __restrict__ wt,
              int Ktot, int Ntot)
{
  __shared__ float tile[32][33];
  const int tid = threadIdx.x;
  const int N0 = blockIdx.x * 32, K0 = blockIdx.y * 32;
  const int r = tid >> 3, c4 = (tid & 7) * 4;
  const float4 f = *(const float4*)(w + (size_t)(K0 + r) * Ntot + N0 + c4);
  tile[r][c4 + 0] = f.x; tile[r][c4 + 1] = f.y;
  tile[r][c4 + 2] = f.z; tile[r][c4 + 3] = f.w;
  __syncthreads();
  bf16x4 b4;
#pragma unroll
  for (int j = 0; j < 4; ++j) b4[j] = (__bf16)tile[c4 + j][r];
  *(bf16x4*)(wt + (size_t)(N0 + r) * Ktot + K0 + c4) = b4;
}

// ---------------------------------------------------------------------------
// GEMM: C = A(bf16 [M][1024]) x Bt(bf16 [N][1024])^T, f32 accumulate.
// BM=BN=128, BK=32; 8 waves (2M x 4N), wave = 64x32 patch = 4x2 WMMA tiles.
// TDM double-buffered staging (LDS rows padded to 40 elems = 16DW + 4DW pad).
// MODE 0: +b_qkv, scatter to Qh (x0.125)/Kh/Vt.  MODE 1: +b_out -> fp32 out.
// ---------------------------------------------------------------------------
template <int MODE>
__global__ void __launch_bounds__(256)
gemm_bf16_kernel(const __bf16* __restrict__ A, const __bf16* __restrict__ Bt,
                 const float* __restrict__ bias,
                 __bf16* __restrict__ Qh, __bf16* __restrict__ Kh,
                 __bf16* __restrict__ Vt, float* __restrict__ out)
{
  const int Ktot = 1024;
  __shared__ __bf16 As[2][128 * 40];
  __shared__ __bf16 Bs[2][128 * 40];

  const int tid  = threadIdx.x;
  const int lane = tid & 31;
  const int wv   = tid >> 5;
  const int wm   = wv & 1;
  const int wn   = wv >> 1;
  const int hi   = lane >> 4;
  const int lo16 = lane & 15;
  const int M0   = blockIdx.y * 128;
  const int N0   = blockIdx.x * 128;

  const __bf16* gA = A + (size_t)M0 * Ktot;
  const __bf16* gB = Bt + (size_t)N0 * Ktot;

  auto stage = [&](int it, int buf) {
#if HAVE_TDM
    if (tid < 32) {  // wave 0 issues both DMAs
      tdm_load_2d(lds_lo32(&As[buf][0]), gA + 32 * it, 32, 128, Ktot,
                  Ktot, M_TOK, 3, 3);
      tdm_load_2d(lds_lo32(&Bs[buf][0]), gB + 32 * it, 32, 128, Ktot,
                  Ktot, (MODE == 0) ? 3072 : 1024, 3, 3);
    }
#else
    const int row = tid >> 1, cg = tid & 1;
    {
      const uint4* s = (const uint4*)(gA + 32 * it + (size_t)row * Ktot + 16 * cg);
      uint4 u0 = s[0], u1 = s[1];
      *(uint4*)&As[buf][row * 40 + 16 * cg]     = u0;
      *(uint4*)&As[buf][row * 40 + 16 * cg + 8] = u1;
    }
    {
      const uint4* s = (const uint4*)(gB + 32 * it + (size_t)row * Ktot + 16 * cg);
      uint4 u0 = s[0], u1 = s[1];
      *(uint4*)&Bs[buf][row * 40 + 16 * cg]     = u0;
      *(uint4*)&Bs[buf][row * 40 + 16 * cg + 8] = u1;
    }
#endif
  };

  f32x8 acc[4][2];
#pragma unroll
  for (int mi = 0; mi < 4; ++mi)
#pragma unroll
    for (int ni = 0; ni < 2; ++ni) acc[mi][ni] = zero8();

  const int iters = Ktot / 32;
  stage(0, 0);
  for (int it = 0; it < iters; ++it) {
    const int buf = it & 1;
    if (it + 1 < iters) { stage(it + 1, buf ^ 1); TDM_WAIT(2); }
    else                { TDM_WAIT(0); }
    __syncthreads();

    AFragU a[4], b[2];
#pragma unroll
    for (int mi = 0; mi < 4; ++mi) {
      const __bf16* base = &As[buf][(wm * 64 + mi * 16 + lo16) * 40];
      a[mi].h[0] = *(const bf16x8*)(base + 8 * hi);
      a[mi].h[1] = *(const bf16x8*)(base + 16 + 8 * hi);
    }
#pragma unroll
    for (int ni = 0; ni < 2; ++ni) {
      const __bf16* base = &Bs[buf][(wn * 32 + ni * 16 + lo16) * 40];
      b[ni].h[0] = *(const bf16x8*)(base + 8 * hi);
      b[ni].h[1] = *(const bf16x8*)(base + 16 + 8 * hi);
    }
#pragma unroll
    for (int mi = 0; mi < 4; ++mi)
#pragma unroll
      for (int ni = 0; ni < 2; ++ni)
        acc[mi][ni] = wmma_bf16(a[mi].v, b[ni].v, acc[mi][ni]);
    __syncthreads();
  }

  // epilogue: D element (r,lane): M = base + r + 8*hi, N = base + (lane&15)
#pragma unroll
  for (int mi = 0; mi < 4; ++mi)
#pragma unroll
    for (int ni = 0; ni < 2; ++ni)
#pragma unroll
      for (int r = 0; r < 8; ++r) {
        const int row = M0 + wm * 64 + mi * 16 + r + 8 * hi;
        const int col = N0 + wn * 32 + ni * 16 + lo16;
        const float v = acc[mi][ni][r] + bias[col];
        if constexpr (MODE == 0) {
          const int sec = col >> 10;  // 0=Q,1=K,2=V (uniform per block)
          const int c = col & 1023;
          const int h = c >> 6, d = c & 63;
          const int bb = row >> 11, t = row & (T_SEQ - 1);
          const int bh = bb * N_HEAD + h;
          if (sec == 0)
            Qh[((size_t)bh * T_SEQ + t) * HEAD_DIM + d] = (__bf16)(v * 0.125f);
          else if (sec == 1)
            Kh[((size_t)bh * T_SEQ + t) * HEAD_DIM + d] = (__bf16)v;
          else
            Vt[((size_t)bh * HEAD_DIM + d) * T_SEQ + t] = (__bf16)v;
        } else {
          out[(size_t)row * 1024 + col] = v;
        }
      }
}

// ---------------------------------------------------------------------------
// Kernel 2: transposed flash attention. Block = 4 waves, 64 queries; wave owns
// 16 queries. 32-key chunks TDM-staged, double-buffered.
//   S^T (16k x 16q) = Kc(16x64) x Q^T(64x16);  O^T (d x q) = V^T x P^T.
// Softmax stats live per-lane (query = lane&15).
// ---------------------------------------------------------------------------
__global__ void __launch_bounds__(128)
attn_kernel(const __bf16* __restrict__ Qh, const __bf16* __restrict__ Kh,
            const __bf16* __restrict__ Vt, __bf16* __restrict__ Ym)
{
  __shared__ __bf16 Ks[2][32 * 72];   // [key][d], 128B row + 16B pad
  __shared__ __bf16 Vts[2][64 * 40];  // [d][key], 64B row + 16B pad

  const int tid  = threadIdx.x;
  const int lane = tid & 31;
  const int wv   = tid >> 5;
  const int hi   = lane >> 4;
  const int lo16 = lane & 15;
  const int blk  = blockIdx.x;
  const int qt   = blk & 31;
  const int bh   = blk >> 5;
  const int qbase = qt * 64;
  const int qw    = qbase + wv * 16;

  const __bf16* Qp = Qh + (size_t)bh * T_SEQ * HEAD_DIM;
  const __bf16* Kp = Kh + (size_t)bh * T_SEQ * HEAD_DIM;
  const __bf16* Vp = Vt + (size_t)bh * HEAD_DIM * T_SEQ;

  auto stage = [&](int ch, int buf) {
#if HAVE_TDM
    if (wv == 0) {
      tdm_load_2d(lds_lo32(&Ks[buf][0]), Kp + (size_t)(ch * 32) * HEAD_DIM,
                  64, 32, 64, 64, T_SEQ, 4, 3);
      tdm_load_2d(lds_lo32(&Vts[buf][0]), Vp + (size_t)(ch * 32),
                  32, 64, T_SEQ, T_SEQ, 64, 3, 3);
    }
#else
    const int kb = ch * 32;
    {
      const int row = tid >> 2, cg = tid & 3;
      const uint4* s = (const uint4*)(Kp + (size_t)(kb + row) * HEAD_DIM + cg * 16);
      uint4 u0 = s[0], u1 = s[1];
      *(uint4*)&Ks[buf][row * 72 + cg * 16]     = u0;
      *(uint4*)&Ks[buf][row * 72 + cg * 16 + 8] = u1;
    }
    {
      const int row = tid >> 1, cg = tid & 1;
      const uint4* s = (const uint4*)(Vp + (size_t)row * T_SEQ + kb + cg * 16);
      uint4 u0 = s[0], u1 = s[1];
      *(uint4*)&Vts[buf][row * 40 + cg * 16]     = u0;
      *(uint4*)&Vts[buf][row * 40 + cg * 16 + 8] = u1;
    }
#endif
  };

  // persistent Q^T B-fragments
  AFragU bQ[2];
  {
    const __bf16* qrow = Qp + (size_t)(qw + lo16) * HEAD_DIM;
#pragma unroll
    for (int kc = 0; kc < 2; ++kc) {
      bQ[kc].h[0] = *(const bf16x8*)(qrow + 32 * kc + 8 * hi);
      bQ[kc].h[1] = *(const bf16x8*)(qrow + 32 * kc + 16 + 8 * hi);
    }
  }

  f32x8 o[4];
#pragma unroll
  for (int dt = 0; dt < 4; ++dt) o[dt] = zero8();
  float m_run = -INFINITY, l_run = 0.0f;

  const int nch = (qbase + 64) >> 5;
  stage(0, 0);
  for (int ch = 0; ch < nch; ++ch) {
    const int buf = ch & 1;
    const int kbase = ch * 32;
    if (ch + 1 < nch) { stage(ch + 1, buf ^ 1); TDM_WAIT(2); }
    else              { TDM_WAIT(0); }
    __syncthreads();

    if (kbase <= qw + 15) {            // wave-uniform -> EXEC stays full
      f32x8 s[2];
      s[0] = zero8(); s[1] = zero8();
#pragma unroll
      for (int t = 0; t < 2; ++t) {
        const __bf16* krow = &Ks[buf][(t * 16 + lo16) * 72];
#pragma unroll
        for (int kc = 0; kc < 2; ++kc) {
          AFragU aK;
          aK.h[0] = *(const bf16x8*)(krow + kc * 32 + 8 * hi);
          aK.h[1] = *(const bf16x8*)(krow + kc * 32 + 16 + 8 * hi);
          s[t] = wmma_bf16(aK.v, bQ[kc].v, s[t]);
        }
      }
      const int q = qw + lo16;
      if (kbase + 31 > qw) {           // causal mask (diagonal chunks only)
#pragma unroll
        for (int t = 0; t < 2; ++t)
#pragma unroll
          for (int r = 0; r < 8; ++r) {
            const int key = kbase + t * 16 + 8 * hi + r;
            if (key > q) s[t][r] = -INFINITY;
          }
      }
      // online softmax (query lives in lane&15; halves merged via shfl_xor 16)
      float mloc = -INFINITY;
#pragma unroll
      for (int t = 0; t < 2; ++t)
#pragma unroll
        for (int r = 0; r < 8; ++r) mloc = fmaxf(mloc, s[t][r]);
      mloc = fmaxf(mloc, __shfl_xor(mloc, 16, 32));
      const float mnew  = fmaxf(m_run, mloc);
      const float alpha = __expf(m_run - mnew);
      float p0[8], p1[8], ls = 0.0f;
#pragma unroll
      for (int r = 0; r < 8; ++r) {
        p0[r] = __expf(s[0][r] - mnew);
        p1[r] = __expf(s[1][r] - mnew);
        ls += p0[r] + p1[r];
      }
      ls += __shfl_xor(ls, 16, 32);
      l_run = l_run * alpha + ls;
      m_run = mnew;
#pragma unroll
      for (int dt = 0; dt < 4; ++dt)
#pragma unroll
        for (int r = 0; r < 8; ++r) o[dt][r] *= alpha;

      // P^T B-fragment from S^T accum layout (half-swap via shfl_xor 16)
      AFragU bP;
      const bool lolane = (hi == 0);
#pragma unroll
      for (int r = 0; r < 8; ++r) {
        const float x0 = __shfl_xor(p0[r], 16, 32);
        const float x1 = __shfl_xor(p1[r], 16, 32);
        bP.v[r]     = (__bf16)(lolane ? p0[r] : x1);
        bP.v[r + 8] = (__bf16)(lolane ? x0 : p1[r]);
      }
      // O^T += V^T * P^T
#pragma unroll
      for (int dt = 0; dt < 4; ++dt) {
        const __bf16* vrow = &Vts[buf][(dt * 16 + lo16) * 40];
        AFragU aV;
        aV.h[0] = *(const bf16x8*)(vrow + 16 * hi);
        aV.h[1] = *(const bf16x8*)(vrow + 16 * hi + 8);
        o[dt] = wmma_bf16(aV.v, bP.v, o[dt]);
      }
    }
    __syncthreads();
  }

  const float rl = 1.0f / l_run;
  const int b = bh >> 4, h = bh & 15;
  const int q = qw + lo16;
  __bf16* yrow = Ym + (size_t)(b * T_SEQ + q) * D_MODEL + h * HEAD_DIM;
#pragma unroll
  for (int dt = 0; dt < 4; ++dt)
#pragma unroll
    for (int r = 0; r < 8; ++r) {
      const int d = dt * 16 + 8 * hi + r;
      yrow[d] = (__bf16)(o[dt][r] * rl);
    }
}

// ---------------------------------------------------------------------------
extern "C" void kernel_launch(void* const* d_in, const int* in_sizes, int n_in,
                              void* d_out, int out_size, void* d_ws, size_t ws_size,
                              hipStream_t stream) {
  (void)in_sizes; (void)n_in; (void)out_size; (void)ws_size;
  const float* x     = (const float*)d_in[0];
  const float* w_qkv = (const float*)d_in[1];
  const float* b_qkv = (const float*)d_in[2];
  const float* w_out = (const float*)d_in[3];
  const float* b_out = (const float*)d_in[4];
  float* out = (float*)d_out;

  // workspace (bf16): Qh,Kh,Vt,Ym (4M elems each), xb (4M), Wqkvt (3M), Woutt (1M)
  const size_t szQ = (size_t)BATCH * N_HEAD * T_SEQ * HEAD_DIM;  // 4M
  __bf16* Qh    = (__bf16*)d_ws;
  __bf16* Kh    = Qh + szQ;
  __bf16* Vt    = Kh + szQ;
  __bf16* Ym    = Vt + szQ;
  __bf16* xb    = Ym + (size_t)M_TOK * D_MODEL;
  __bf16* Wqkvt = xb + (size_t)M_TOK * D_MODEL;
  __bf16* Woutt = Wqkvt + (size_t)D_MODEL * 3 * D_MODEL;         // total 48 MB

  cvt_x_kernel<<<dim3((M_TOK * D_MODEL) / 1024), 256, 0, stream>>>(x, xb);
  wtrans_kernel<<<dim3(3072 / 32, 1024 / 32), 256, 0, stream>>>(w_qkv, Wqkvt,
                                                                1024, 3072);
  wtrans_kernel<<<dim3(1024 / 32, 1024 / 32), 256, 0, stream>>>(w_out, Woutt,
                                                                1024, 1024);
  gemm_bf16_kernel<0><<<dim3(3072 / 128, M_TOK / 128), 256, 0, stream>>>(
      xb, Wqkvt, b_qkv, Qh, Kh, Vt, nullptr);
  attn_kernel<<<dim3(BATCH * N_HEAD * (T_SEQ / 64)), 128, 0, stream>>>(
      Qh, Kh, Vt, Ym);
  gemm_bf16_kernel<1><<<dim3(1024 / 128, M_TOK / 128), 256, 0, stream>>>(
      Ym, Woutt, b_out, nullptr, nullptr, nullptr, out);
}
